// BitwiseTasNetBlock_77017353552488
// MI455X (gfx1250) — compile-verified, matrix operand order checked by reading, not run on hardware
//
#include <hip/hip_runtime.h>
#include <hip/hip_bf16.h>

// ---------------- problem constants ----------------
#define B_SZ 8
#define C_SZ 128
#define D_SZ 512
#define T_LEN 8000
#define L_SZ 4
#define K_SZ 3
#define EPS_BN 1e-5f

// ---------------- WMMA / vector types ----------------
typedef __attribute__((ext_vector_type(16))) __bf16 v16bf;
typedef __attribute__((ext_vector_type(8)))  float  v8f;
typedef __attribute__((ext_vector_type(4)))  unsigned int  u32x4;
typedef __attribute__((ext_vector_type(8)))  int           i32x8;
typedef __attribute__((ext_vector_type(4)))  int           i32x4;
typedef __attribute__((ext_vector_type(4)))  unsigned short us4;

__device__ __forceinline__ unsigned short f2bf(float f) {
  union { float f; unsigned u; } c; c.f = f;
  unsigned u = c.u;
  unsigned r = (u + 0x7FFFu + ((u >> 16) & 1u)) >> 16;  // RNE
  return (unsigned short)r;
}
__device__ __forceinline__ float bf2f(unsigned short h) {
  union { unsigned u; float f; } c; c.u = ((unsigned)h) << 16;
  return c.f;
}

// ---------------- GEMM tile config ----------------
#define MTILE 64
#define NTILE 128
#define KCHUNK 128
#define LDAP_A 130          // A row stride in halves: 256B data + 1 DWORD TDM pad = 260B
#define LDAP_B (KCHUNK + 8) // B (transposed) row stride in halves

#if defined(__has_builtin)
#if __has_builtin(__builtin_amdgcn_tensor_load_to_lds)
#define HAVE_TDM 1
#endif
#endif

#ifdef HAVE_TDM
// Issue a 2D TDM tile load: global (row-major, stride0 elems) -> LDS (+1 DWORD pad / 64 DWORDs).
// data_size = 2 bytes. tile_d0 elements per row (contiguous), tile_d1 rows.
__device__ __forceinline__ void tdm_load_2d_bf16(
    unsigned lds_off, const void* gaddr,
    unsigned tile_d0, unsigned tile_d1,
    unsigned long long stride0_elems)
{
  const unsigned long long ga = (unsigned long long)gaddr;
  u32x4 g0;
  g0[0] = 1u;                                            // count=1, user mode
  g0[1] = lds_off;                                       // lds_addr (bytes)
  g0[2] = (unsigned)(ga & 0xFFFFFFFFu);                  // global_addr lo
  g0[3] = (unsigned)((ga >> 32) & 0x01FFFFFFu) | (2u << 30);  // addr[56:32] | type=2
  const unsigned tensor_d0 = tile_d0, tensor_d1 = tile_d1;
  i32x8 g1;
  g1[0] = (int)((1u << 16)        // data_size = 2 bytes
              | (1u << 20)        // pad_enable
              | (5u << 22)        // pad_interval: 64 DWORDs (one 256B row)
              | (0u << 25));      // pad_amount: 1 DWORD
  g1[1] = (int)((tensor_d0 & 0xFFFFu) << 16);                         // dim0 lo16
  g1[2] = (int)(((tensor_d0 >> 16) & 0xFFFFu) | ((tensor_d1 & 0xFFFFu) << 16));
  g1[3] = (int)(((tensor_d1 >> 16) & 0xFFFFu) | ((tile_d0 & 0xFFFFu) << 16));
  g1[4] = (int)(tile_d1 & 0xFFFFu);                                   // tile_dim1, tile_dim2=0
  g1[5] = (int)(unsigned)(stride0_elems & 0xFFFFFFFFull);             // dim0_stride lo32
  g1[6] = (int)(unsigned)((stride0_elems >> 32) & 0xFFFFull);         // dim0_stride hi16
  g1[7] = 0;
  const i32x4 z4 = {0, 0, 0, 0};
#if __clang_major__ >= 23
  const i32x8 z8 = {0, 0, 0, 0, 0, 0, 0, 0};
  __builtin_amdgcn_tensor_load_to_lds(g0, g1, z4, z4, z8, 0);
#else
  __builtin_amdgcn_tensor_load_to_lds(g0, g1, z4, z4, 0);
#endif
}
#endif

// out[b,m,t] = prelu( sum_k W[m,k]*X[b,k,t] + bias[m] ), optional stats, optional residual.
// X bf16 [B,Ktot,T]; W bf16 [M,Ktot]; Out [B,M,T] (bf16, or fp32 + resid).
__global__ __launch_bounds__(256) void k_gemm_bf16(
    const unsigned short* __restrict__ Xb,
    const unsigned short* __restrict__ W,
    const float* __restrict__ bias,
    const float* __restrict__ resid,           // null unless final fp32 output
    void* __restrict__ Out, int out_is_f32,
    int M, int Ktot,
    const float* __restrict__ aptr, int do_prelu,
    float* __restrict__ gsum, float* __restrict__ gsumsq, int do_stats)
{
  __shared__ unsigned short ldsA[MTILE][LDAP_A];   // W tile [m][k] (TDM-written, padded rows)
  __shared__ unsigned short ldsB[NTILE][LDAP_B];   // X tile transposed [n][k]
  __shared__ float sSum[MTILE];
  __shared__ float sSq[MTILE];

  const int tid  = threadIdx.x;
  const int b    = blockIdx.z;
  const int m0   = blockIdx.y * MTILE;
  const int n0   = blockIdx.x * NTILE;
  const int wave = tid >> 5;
  const int lane = tid & 31;
  const int mw   = wave & 1;    // wave's 32-row sub-tile (0..1)
  const int nw   = wave >> 1;   // wave's 32-col sub-tile (0..3)
  const int hi   = lane >> 4;   // lane half (K split per ISA layout)
  const int ll   = lane & 15;

  const float alpha = do_prelu ? aptr[0] : 0.0f;

  if (tid < MTILE) { sSum[tid] = 0.0f; sSq[tid] = 0.0f; }

  v8f acc[2][2];
  acc[0][0] = (v8f){}; acc[0][1] = (v8f){};
  acc[1][0] = (v8f){}; acc[1][1] = (v8f){};

  const size_t xbase = (size_t)b * (size_t)Ktot * T_LEN;

  // B staging mapping: 4 consecutive t-columns per thread, 8 k-rows covered per pass
  const int nq4 = (lane == lane ? (tid & 31) * 4 : 0);
  const int kr0 = tid >> 5;                 // 0..7
  const int tq  = n0 + nq4;
  const bool qv = (tq + 3) < T_LEN;         // T and n0 are multiples of 4: all-or-nothing

  for (int kb = 0; kb < Ktot; kb += KCHUNK) {
#ifdef HAVE_TDM
    // ---- async DMA of the 64x128 bf16 weight tile straight into LDS ----
    if (tid == 0) {
      tdm_load_2d_bf16((unsigned)(size_t)&ldsA[0][0],
                       W + (size_t)m0 * Ktot + kb,
                       KCHUNK, MTILE, (unsigned long long)Ktot);
    }
#else
    for (int it = 0; it < 8; ++it) {
      int idx = it * 256 + tid;
      int r   = idx >> 5;
      int c4  = (idx & 31) * 4;
      *(uint2*)&ldsA[r][c4] = *(const uint2*)(W + (size_t)(m0 + r) * Ktot + kb + c4);
    }
#endif
    // ---- stage B transposed: X[k][t] -> ldsB[n][k], 8B vector global loads ----
    for (int it = 0; it < KCHUNK / 8; ++it) {
      const int k = kr0 + it * 8;
      us4 v = {0, 0, 0, 0};
      if (qv) v = *(const us4*)(Xb + xbase + (size_t)(kb + k) * T_LEN + tq);
      ldsB[nq4 + 0][k] = v.x;
      ldsB[nq4 + 1][k] = v.y;
      ldsB[nq4 + 2][k] = v.z;
      ldsB[nq4 + 3][k] = v.w;
    }
    if (qv && (kb + KCHUNK < Ktot))
      __builtin_prefetch(Xb + xbase + (size_t)(kb + KCHUNK + kr0) * T_LEN + tq, 0, 1);
#ifdef HAVE_TDM
    if (tid == 0) __builtin_amdgcn_s_wait_tensorcnt(0);
#endif
    __syncthreads();

    // ---- 4 WMMA K-steps of 32 ----
    for (int ks = 0; ks < 4; ++ks) {
      const int kbase = ks * 32;
      union Frag { v16bf v; unsigned u[8]; } aF[2], bF[2];
      for (int s = 0; s < 2; ++s) {
        const int mr = mw * 32 + s * 16 + ll;
        const int nc = nw * 32 + s * 16 + ll;
        #pragma unroll
        for (int j = 0; j < 8; ++j) {
          // 16-bit A-layout (16x32): VGPR j holds a K pair; lanes 16-31 offset K by 8
          const int kk = kbase + ((j & 3) * 2) + ((j >> 2) * 16) + hi * 8;
          aF[s].u[j] = *(const unsigned*)&ldsA[mr][kk];
          bF[s].u[j] = *(const unsigned*)&ldsB[nc][kk];
        }
      }
      acc[0][0] = __builtin_amdgcn_wmma_f32_16x16x32_bf16(false, aF[0].v, false, bF[0].v, (short)0, acc[0][0], false, false);
      acc[0][1] = __builtin_amdgcn_wmma_f32_16x16x32_bf16(false, aF[0].v, false, bF[1].v, (short)0, acc[0][1], false, false);
      acc[1][0] = __builtin_amdgcn_wmma_f32_16x16x32_bf16(false, aF[1].v, false, bF[0].v, (short)0, acc[1][0], false, false);
      acc[1][1] = __builtin_amdgcn_wmma_f32_16x16x32_bf16(false, aF[1].v, false, bF[1].v, (short)0, acc[1][1], false, false);
    }
    __syncthreads();
  }

  // ---- fused epilogue: bias + PReLU + store + BN stats ----
  for (int s = 0; s < 2; ++s) {
    for (int u = 0; u < 2; ++u) {
      v8f a = acc[s][u];
      const int t = n0 + nw * 32 + u * 16 + ll;
      const bool tv = t < T_LEN;
      #pragma unroll
      for (int r = 0; r < 8; ++r) {
        const int mloc = mw * 32 + s * 16 + hi * 8 + r;   // C/D layout: VGPR r -> M=r(+8)
        const int m = m0 + mloc;
        float v = a[r] + bias[m];
        if (do_prelu) v = v >= 0.0f ? v : alpha * v;
        if (tv) {
          const size_t oi = (size_t)b * M * T_LEN + (size_t)m * T_LEN + t;
          if (out_is_f32) {
            float o = v;
            if (resid) o += resid[oi];
            ((float*)Out)[oi] = o;
          } else {
            ((unsigned short*)Out)[oi] = f2bf(v);
          }
        }
        if (do_stats) {
          float sv = tv ? v : 0.0f;
          float sq = sv * sv;
          #pragma unroll
          for (int off = 8; off >= 1; off >>= 1) {    // reduce across N (16 lanes)
            sv += __shfl_xor(sv, off, 16);
            sq += __shfl_xor(sq, off, 16);
          }
          if (ll == 0) {
            atomicAdd(&sSum[mloc], sv);
            atomicAdd(&sSq[mloc], sq);
          }
        }
      }
    }
  }
  if (do_stats) {
    __syncthreads();
    if (tid < MTILE) {
      atomicAdd(&gsum[m0 + tid], sSum[tid]);
      atomicAdd(&gsumsq[m0 + tid], sSq[tid]);
    }
  }
}

// ---- dilated depthwise conv: BN1 affine -> conv K=3 -> bias -> PReLU -> bf16 + stats ----
__global__ __launch_bounds__(256) void k_depthwise(
    const unsigned short* __restrict__ p1,
    const float* __restrict__ scale1, const float* __restrict__ shift1,
    const float* __restrict__ wd, const float* __restrict__ bd,
    const float* __restrict__ aptr, int dil,
    unsigned short* __restrict__ p2,
    float* __restrict__ gsum, float* __restrict__ gsumsq)
{
  const int d = blockIdx.y, b = blockIdx.z, chunk = blockIdx.x;
  const size_t rb = ((size_t)b * D_SZ + d) * T_LEN;
  const float sc = scale1[d], sh = shift1[d];
  const float w0 = wd[d * K_SZ + 0], w1 = wd[d * K_SZ + 1], w2 = wd[d * K_SZ + 2];
  const float bias = bd[d];
  const float alpha = aptr[0];

  float s = 0.0f, ss = 0.0f;
  for (int it = 0; it < 8; ++it) {
    const int t = chunk * 2048 + it * 256 + (int)threadIdx.x;
    if (t < T_LEN) {
      float xc = bf2f(p1[rb + t]) * sc + sh;
      float xl = (t - dil >= 0)    ? bf2f(p1[rb + t - dil]) * sc + sh : 0.0f;
      float xr = (t + dil < T_LEN) ? bf2f(p1[rb + t + dil]) * sc + sh : 0.0f;
      float y = w0 * xl + w1 * xc + w2 * xr + bias;
      y = y >= 0.0f ? y : alpha * y;
      p2[rb + t] = f2bf(y);
      s += y; ss += y * y;
    }
  }
  #pragma unroll
  for (int off = 16; off >= 1; off >>= 1) {
    s  += __shfl_xor(s,  off, 32);
    ss += __shfl_xor(ss, off, 32);
  }
  __shared__ float rs[8], rss[8];
  const int wv = threadIdx.x >> 5;
  if ((threadIdx.x & 31) == 0) { rs[wv] = s; rss[wv] = ss; }
  __syncthreads();
  if (threadIdx.x == 0) {
    float ts = 0.0f, tss = 0.0f;
    for (int w = 0; w < 8; ++w) { ts += rs[w]; tss += rss[w]; }
    atomicAdd(&gsum[d], ts);
    atomicAdd(&gsumsq[d], tss);
  }
}

// ---- BN finalize: stats -> per-channel scale/shift ----
__global__ void k_bn_finalize(const float* __restrict__ sum, const float* __restrict__ sumsq,
                              const float* __restrict__ g, const float* __restrict__ be,
                              float* __restrict__ scale, float* __restrict__ shift,
                              int n, float invN)
{
  const int i = blockIdx.x * blockDim.x + threadIdx.x;
  if (i < n) {
    const float mean = sum[i] * invN;
    const float var  = sumsq[i] * invN - mean * mean;
    const float rs   = rsqrtf(var + EPS_BN);
    const float sc   = g[i] * rs;
    scale[i] = sc;
    shift[i] = be[i] - mean * sc;
  }
}

// ---- fold BN2 into w2 / b2: w2b = bf16(w2*scale2), bias2' = b2 + sum_d w2*shift2 ----
__global__ __launch_bounds__(256) void k_fold_w2(
    const float* __restrict__ w2, const float* __restrict__ scale2,
    const float* __restrict__ shift2, const float* __restrict__ b2,
    unsigned short* __restrict__ w2b, float* __restrict__ bias2p)
{
  const int c = blockIdx.x;
  float pb = 0.0f;
  for (int d = threadIdx.x; d < D_SZ; d += 256) {
    const float w = w2[(size_t)c * D_SZ + d];
    w2b[(size_t)c * D_SZ + d] = f2bf(w * scale2[d]);
    pb += w * shift2[d];
  }
  #pragma unroll
  for (int off = 16; off >= 1; off >>= 1) pb += __shfl_xor(pb, off, 32);
  __shared__ float r[8];
  const int wv = threadIdx.x >> 5;
  if ((threadIdx.x & 31) == 0) r[wv] = pb;
  __syncthreads();
  if (threadIdx.x == 0) {
    float t = 0.0f;
    for (int w = 0; w < 8; ++w) t += r[w];
    bias2p[c] = b2[c] + t;
  }
}

__global__ void k_cvt_bf16(const float* __restrict__ in, unsigned short* __restrict__ out, int n) {
  const int i = blockIdx.x * blockDim.x + threadIdx.x;
  if (i < n) out[i] = f2bf(in[i]);
}

__global__ void k_zero(float* __restrict__ p, int n) {
  const int i = blockIdx.x * blockDim.x + threadIdx.x;
  if (i < n) p[i] = 0.0f;
}

// ---------------- host-side orchestration ----------------
extern "C" void kernel_launch(void* const* d_in, const int* in_sizes, int n_in,
                              void* d_out, int out_size, void* d_ws, size_t ws_size,
                              hipStream_t stream) {
  const float* x   = (const float*)d_in[0];
  const float* w1  = (const float*)d_in[1];
  const float* b1  = (const float*)d_in[2];
  const float* a1  = (const float*)d_in[3];
  const float* g1p = (const float*)d_in[4];
  const float* be1 = (const float*)d_in[5];
  const float* wd  = (const float*)d_in[6];
  const float* bd  = (const float*)d_in[7];
  const float* a2  = (const float*)d_in[8];
  const float* g2p = (const float*)d_in[9];
  const float* be2 = (const float*)d_in[10];
  const float* w2  = (const float*)d_in[11];
  const float* b2  = (const float*)d_in[12];

  // carve workspace (256B aligned slices)
  char* ws = (char*)d_ws;
  size_t off = 0;
  auto carve = [&](size_t bytes) -> void* {
    void* p = ws + off;
    off = (off + bytes + 255) & ~(size_t)255;
    return p;
  };
  unsigned short* xb   = (unsigned short*)carve((size_t)B_SZ * C_SZ * T_LEN * 2);
  unsigned short* hbuf = (unsigned short*)carve((size_t)B_SZ * C_SZ * T_LEN * 2);
  unsigned short* p1   = (unsigned short*)carve((size_t)B_SZ * D_SZ * T_LEN * 2);
  unsigned short* p2   = (unsigned short*)carve((size_t)B_SZ * D_SZ * T_LEN * 2);
  unsigned short* w1b  = (unsigned short*)carve((size_t)D_SZ * C_SZ * 2);
  unsigned short* w2b  = (unsigned short*)carve((size_t)C_SZ * D_SZ * 2);
  float* bias2p = (float*)carve(C_SZ * 4);
  float* stats  = (float*)carve(4 * D_SZ * 4);     // sum1, sq1, sum2, sq2
  float* sum1 = stats, *sq1 = stats + D_SZ, *sum2 = stats + 2 * D_SZ, *sq2 = stats + 3 * D_SZ;
  float* scale1 = (float*)carve(D_SZ * 4);
  float* shift1 = (float*)carve(D_SZ * 4);
  float* scale2 = (float*)carve(D_SZ * 4);
  float* shift2 = (float*)carve(D_SZ * 4);

  const float invN = 1.0f / (float)(B_SZ * T_LEN);
  const dim3 blk(256);
  const dim3 grid_g1((T_LEN + NTILE - 1) / NTILE, D_SZ / MTILE, B_SZ);  // 63 x 8 x 8
  const dim3 grid_g2((T_LEN + NTILE - 1) / NTILE, C_SZ / MTILE, B_SZ);  // 63 x 2 x 8
  const dim3 grid_dw(4, D_SZ, B_SZ);

  // one-time fp32 -> bf16 conversion of the input
  const int nX = B_SZ * C_SZ * T_LEN;
  k_cvt_bf16<<<(nX + 255) / 256, blk, 0, stream>>>(x, xb, nX);

  for (int i = 0; i < L_SZ; ++i) {
    k_zero<<<(4 * D_SZ + 255) / 256, blk, 0, stream>>>(stats, 4 * D_SZ);
    k_cvt_bf16<<<(D_SZ * C_SZ + 255) / 256, blk, 0, stream>>>(w1 + (size_t)i * D_SZ * C_SZ, w1b, D_SZ * C_SZ);

    // 1x1 conv C->D (+bias, PReLU, BN1 stats), bf16 out
    k_gemm_bf16<<<grid_g1, blk, 0, stream>>>(
        (i == 0) ? xb : hbuf,
        w1b, b1 + i * D_SZ, nullptr, (void*)p1, 0,
        D_SZ, C_SZ, a1 + i, 1, sum1, sq1, 1);

    k_bn_finalize<<<(D_SZ + 255) / 256, blk, 0, stream>>>(sum1, sq1, g1p + i * D_SZ, be1 + i * D_SZ,
                                                          scale1, shift1, D_SZ, invN);

    // dilated depthwise conv (+BN1 affine, bias, PReLU, BN2 stats)
    k_depthwise<<<grid_dw, blk, 0, stream>>>(p1, scale1, shift1,
                                             wd + (size_t)i * D_SZ * K_SZ, bd + i * D_SZ,
                                             a2 + i, 1 << i, p2, sum2, sq2);

    k_bn_finalize<<<(D_SZ + 255) / 256, blk, 0, stream>>>(sum2, sq2, g2p + i * D_SZ, be2 + i * D_SZ,
                                                          scale2, shift2, D_SZ, invN);

    // fold BN2 into w2/b2
    k_fold_w2<<<C_SZ, blk, 0, stream>>>(w2 + (size_t)i * C_SZ * D_SZ, scale2, shift2,
                                        b2 + i * C_SZ, w2b, bias2p);

    // 1x1 conv D->C; last layer adds residual and writes fp32 d_out
    const int last = (i == L_SZ - 1);
    k_gemm_bf16<<<grid_g2, blk, 0, stream>>>(
        p2, w2b, bias2p,
        last ? x : nullptr,
        last ? d_out : (void*)hbuf, last,
        C_SZ, D_SZ, nullptr, 0, nullptr, nullptr, 0);
  }
}
